// Head_38603166057048
// MI455X (gfx1250) — compile-verified, matrix operand order checked by reading, not run on hardware
//
#include <hip/hip_runtime.h>
#include <hip/hip_bf16.h>

typedef _Float16 half_t;
typedef __attribute__((ext_vector_type(16))) _Float16 v16h;
typedef __attribute__((ext_vector_type(8)))  float    v8f;

#define N_EMBED   1024
#define HEAD      64
#define TSEQ      4096
#define NBATCH    4
#define NEG_BIG   (-3.0e30f)
#define WPACK_ELEMS 65536   // per matrix: 1024*64

// 16-bit A/B WMMA fragment (16x32) K-mapping:
// element e -> K = (e<8 ? e : e+8) + 8*(lane/16); pair j -> koff(j), koff(j)+1
// inverse:   khalf = (K>>3)&1 ; e = (K&7) + 8*(K>>4)
__device__ __forceinline__ int koff(int j) { return (j < 4) ? (2 * j) : (2 * j + 8); }

__device__ __forceinline__ v8f wmma_f16(v16h a, v16h b, v8f c) {
    return __builtin_amdgcn_wmma_f32_16x16x32_f16(false, a, false, b, (short)0, c, false, false);
}

// ---------------------------------------------------------------------------
// Kernel 0: pack Wq|Wk|Wv (f32 [1024,64]) into f16 B-fragment layout:
// wpack[m][c][n][lane][e]; each lane's 16 halves contiguous (32B).
// ---------------------------------------------------------------------------
__global__ void __launch_bounds__(256)
pack_w_kernel(const float* __restrict__ Wq, const float* __restrict__ Wk,
              const float* __restrict__ Wv, half_t* __restrict__ wpack) {
    const int idx = blockIdx.x * 256 + threadIdx.x;     // 0 .. 3*65536-1
    const int m   = idx >> 16;
    const int r   = idx & (WPACK_ELEMS - 1);
    const int e   = r & 15;
    const int l   = (r >> 4) & 31;
    const int cn  = r >> 9;            // c*4 + n
    const int n   = cn & 3;
    const int c   = cn >> 2;
    const int K   = 32 * c + ((e < 8) ? e : (e + 8)) + 8 * (l >> 4);
    const int col = 16 * n + (l & 15);
    const float* __restrict__ W = (m == 0) ? Wq : (m == 1) ? Wk : Wv;
    wpack[idx] = (half_t)W[(size_t)K * HEAD + col];
}

// ---------------------------------------------------------------------------
// Kernel 1: q,k,v projections. One wave per 16-row tile, full H=64 per wave.
// 12 WMMAs per 32-K chunk. Outputs:
//   qh     f16 [B*T, 64]       pre-scaled by 1/sqrt(H) (for Q B-fragments)
//   khpack f16 [B][T/16][2][32][16]   attention K A-fragment layout
//   vtpack f16 [B][T/32][4][32][16]   attention V^T A-fragment layout
// ---------------------------------------------------------------------------
__global__ void __launch_bounds__(256)
qkv_proj_kernel(const float* __restrict__ x, const half_t* __restrict__ wpack,
                const float* __restrict__ bq, const float* __restrict__ bk,
                const float* __restrict__ bv,
                half_t* __restrict__ qh, half_t* __restrict__ khp,
                half_t* __restrict__ vtp) {
    const int lane   = threadIdx.x & 31;
    const int lane16 = lane & 15;
    const int khalf  = lane >> 4;

    const int rowTile = blockIdx.x * 8 + (threadIdx.x >> 5); // 0 .. B*T/16-1
    const float* __restrict__ xrow =
        x + (size_t)(rowTile * 16 + lane16) * N_EMBED;       // A: M = lane%16

    v8f accq[4] = {}, acck[4] = {}, accv[4] = {};

    for (int c = 0; c < N_EMBED / 32; ++c) {
        // A fragment: x row f32 -> f16 (8 contiguous 8-byte pairs)
        v16h a;
#pragma unroll
        for (int j = 0; j < 8; ++j) {
            const int K  = 32 * c + koff(j) + 8 * khalf;
            a[2 * j]     = (half_t)xrow[K];
            a[2 * j + 1] = (half_t)xrow[K + 1];
        }
        const half_t* __restrict__ wp = wpack + ((size_t)(c * 4) * 32 + lane) * 16;
        v16h bf[12];
#pragma unroll
        for (int n = 0; n < 4; ++n) {
            const half_t* __restrict__ wpn = wp + (size_t)n * 32 * 16;
            bf[3 * n + 0] = *(const v16h*)(wpn);
            bf[3 * n + 1] = *(const v16h*)(wpn + WPACK_ELEMS);
            bf[3 * n + 2] = *(const v16h*)(wpn + 2 * WPACK_ELEMS);
        }
#pragma unroll
        for (int n = 0; n < 4; ++n) {
            accq[n] = wmma_f16(a, bf[3 * n + 0], accq[n]);
            acck[n] = wmma_f16(a, bf[3 * n + 1], acck[n]);
            accv[n] = wmma_f16(a, bf[3 * n + 2], accv[n]);
        }
    }

#pragma unroll
    for (int n = 0; n < 4; ++n) {
        const int col = 16 * n + lane16;   // output column h; C/D: N = lane%16
        const float biasq = bq[col];
        const float biask = bk[col];
        const float biasv = bv[col];
        // k-pack coords from h
        const int kf      = col >> 5;                 // which 32-h fragment
        const int hp      = col & 31;
        const int kh_lane = ((hp >> 3) & 1) * 16;     // + t%16
        const int kh_e    = (hp & 7) + 8 * (hp >> 4);
        // v-pack coords from h
        const int t4      = col >> 4;
        const int v_l16   = col & 15;
#pragma unroll
        for (int r = 0; r < 8; ++r) {
            const int orow = rowTile * 16 + r + 8 * khalf; // M = r + 8*(lane/16)
            const int bidx = orow >> 12;
            const int t    = orow & (TSEQ - 1);
            // q: row major, pre-scaled by 1/sqrt(HEAD)
            qh[(size_t)orow * HEAD + col] = (half_t)((accq[n][r] + biasq) * 0.125f);
            // k: A-fragment layout [b][t/16][kf][lane][e]
            const size_t kidx =
                ((((size_t)bidx * 256 + (t >> 4)) * 2 + kf) * 32 +
                 (kh_lane + (t & 15))) * 16 + kh_e;
            khp[kidx] = (half_t)(acck[n][r] + biask);
            // v: V^T A-fragment layout [b][t/32][t4][lane][e]
            const int Krel = t & 31;
            const size_t vidx =
                ((((size_t)bidx * 128 + (t >> 5)) * 4 + t4) * 32 +
                 (((Krel >> 3) & 1) * 16 + v_l16)) * 16 +
                ((Krel & 7) + 8 * (Krel >> 4));
            vtp[vidx] = (half_t)(accv[n][r] + biasv);
        }
    }
}

// ---------------------------------------------------------------------------
// Kernel 2: flash attention, causal, 64-key chunks. One wave per 16 queries.
// S^T = K * Q^T so softmax'd probabilities convert element-wise into the
// B operands of O^T = V^T * P^T. 16 WMMAs per chunk.
// ---------------------------------------------------------------------------
__global__ void __launch_bounds__(128)
attn_kernel(const half_t* __restrict__ qh, const half_t* __restrict__ khp,
            const half_t* __restrict__ vtp, float* __restrict__ out) {
    const int lane   = threadIdx.x & 31;
    const int lane16 = lane & 15;
    const int khalf  = lane >> 4;

    const int gwave = blockIdx.x * 4 + (threadIdx.x >> 5); // 0..1023
    const int bidx  = gwave >> 8;                          // batch
    const int qbase = (gwave & 255) * 16;                  // query tile start

    // Q^T B-fragments (loaded once): N = query = lane%16, K = h
    const half_t* __restrict__ qrow =
        qh + ((size_t)(bidx << 12) + qbase + lane16) * HEAD;
    v16h bq0, bq1;
#pragma unroll
    for (int j = 0; j < 8; ++j) {
        const int h = koff(j) + 8 * khalf;
        bq0[2 * j]     = qrow[h];
        bq0[2 * j + 1] = qrow[h + 1];
        bq1[2 * j]     = qrow[h + 32];
        bq1[2 * j + 1] = qrow[h + 33];
    }

    float m = NEG_BIG, l = 0.0f;
    v8f o0 = {}, o1 = {}, o2 = {}, o3 = {};

    const int qg = qbase + lane16;  // this lane's query index (global in batch)

    for (int kbase = 0; kbase < qbase + 16; kbase += 64) {
        // --- scores S^T for 64 keys: 4 subtiles x 2 WMMAs ---
        const half_t* __restrict__ kbp =
            khp + ((size_t)(bidx * 256 + (kbase >> 4)) * 1024) + (size_t)lane * 16;
        v8f s0 = {}, s1 = {}, s2 = {}, s3 = {};
        {
            v16h akl = *(const v16h*)(kbp);
            v16h akh = *(const v16h*)(kbp + 512);
            s0 = wmma_f16(akl, bq0, s0);
            s0 = wmma_f16(akh, bq1, s0);
        }
        {
            v16h akl = *(const v16h*)(kbp + 1024);
            v16h akh = *(const v16h*)(kbp + 1536);
            s1 = wmma_f16(akl, bq0, s1);
            s1 = wmma_f16(akh, bq1, s1);
        }
        {
            v16h akl = *(const v16h*)(kbp + 2048);
            v16h akh = *(const v16h*)(kbp + 2560);
            s2 = wmma_f16(akl, bq0, s2);
            s2 = wmma_f16(akh, bq1, s2);
        }
        {
            v16h akl = *(const v16h*)(kbp + 3072);
            v16h akh = *(const v16h*)(kbp + 3584);
            s3 = wmma_f16(akl, bq0, s3);
            s3 = wmma_f16(akh, bq1, s3);
        }

        // --- causal mask (scale already folded into q) ---
        if ((kbase + 63) > qg) {
#pragma unroll
            for (int r = 0; r < 8; ++r) {
                const int kg = kbase + r + 8 * khalf; // key = r + 8*(lane/16)
                if (kg > qg)      s0[r] = NEG_BIG;
                if (kg + 16 > qg) s1[r] = NEG_BIG;
                if (kg + 32 > qg) s2[r] = NEG_BIG;
                if (kg + 48 > qg) s3[r] = NEG_BIG;
            }
        }

        // --- online softmax: one shfl_xor(16) completes each 64-key row ---
        float mc = s0[0];
#pragma unroll
        for (int r = 1; r < 8; ++r) mc = fmaxf(mc, s0[r]);
#pragma unroll
        for (int r = 0; r < 8; ++r) {
            mc = fmaxf(mc, s1[r]);
            mc = fmaxf(mc, s2[r]);
            mc = fmaxf(mc, s3[r]);
        }
        mc = fmaxf(mc, __shfl_xor(mc, 16, 32));

        const float m_new = fmaxf(m, mc);
        const float alpha = __expf(m - m_new);

        float rs = 0.0f;
#pragma unroll
        for (int r = 0; r < 8; ++r) {
            s0[r] = __expf(s0[r] - m_new); rs += s0[r];
            s1[r] = __expf(s1[r] - m_new); rs += s1[r];
            s2[r] = __expf(s2[r] - m_new); rs += s2[r];
            s3[r] = __expf(s3[r] - m_new); rs += s3[r];
        }
        rs += __shfl_xor(rs, 16, 32);
        l = l * alpha + rs;
        m = m_new;

#pragma unroll
        for (int r = 0; r < 8; ++r) {
            o0[r] *= alpha; o1[r] *= alpha; o2[r] *= alpha; o3[r] *= alpha;
        }

        // --- P^T B operands: direct element copy from S^T accumulators ---
        v16h pb0, pb1;
#pragma unroll
        for (int r = 0; r < 8; ++r) {
            pb0[r]     = (half_t)s0[r];  // keys kbase    .. +15
            pb0[r + 8] = (half_t)s1[r];  // keys kbase+16 .. +31
            pb1[r]     = (half_t)s2[r];  // keys kbase+32 .. +47
            pb1[r + 8] = (half_t)s3[r];  // keys kbase+48 .. +63
        }

        // --- O^T += V^T * P^T : 4 h-tiles x 2 key-halves = 8 WMMAs ---
        const half_t* __restrict__ vbp =
            vtp + ((size_t)(bidx * 128 + (kbase >> 5)) * 2048) + (size_t)lane * 16;
        {
            v16h av = *(const v16h*)(vbp);          o0 = wmma_f16(av, pb0, o0);
            v16h aw = *(const v16h*)(vbp + 2048);   o0 = wmma_f16(aw, pb1, o0);
        }
        {
            v16h av = *(const v16h*)(vbp + 512);    o1 = wmma_f16(av, pb0, o1);
            v16h aw = *(const v16h*)(vbp + 2560);   o1 = wmma_f16(aw, pb1, o1);
        }
        {
            v16h av = *(const v16h*)(vbp + 1024);   o2 = wmma_f16(av, pb0, o2);
            v16h aw = *(const v16h*)(vbp + 3072);   o2 = wmma_f16(aw, pb1, o2);
        }
        {
            v16h av = *(const v16h*)(vbp + 1536);   o3 = wmma_f16(av, pb0, o3);
            v16h aw = *(const v16h*)(vbp + 3584);   o3 = wmma_f16(aw, pb1, o3);
        }
    }

    // --- epilogue: O^T element (h = t4*16 + r + 8*khalf, query = lane%16) ---
    const float inv = 1.0f / l;
    float* __restrict__ orow =
        out + ((size_t)(bidx << 12) + qbase + lane16) * HEAD + 8 * khalf;
    v8f oo[4] = {o0, o1, o2, o3};
#pragma unroll
    for (int t4 = 0; t4 < 4; ++t4) {
        float4 lo, hi;
        lo.x = oo[t4][0] * inv; lo.y = oo[t4][1] * inv;
        lo.z = oo[t4][2] * inv; lo.w = oo[t4][3] * inv;
        hi.x = oo[t4][4] * inv; hi.y = oo[t4][5] * inv;
        hi.z = oo[t4][6] * inv; hi.w = oo[t4][7] * inv;
        float4* p = (float4*)(orow + t4 * 16);
        p[0] = lo;
        p[1] = hi;
    }
}

// ---------------------------------------------------------------------------
extern "C" void kernel_launch(void* const* d_in, const int* in_sizes, int n_in,
                              void* d_out, int out_size, void* d_ws, size_t ws_size,
                              hipStream_t stream) {
    const float* x  = (const float*)d_in[0];
    const float* Wq = (const float*)d_in[1];
    const float* bq = (const float*)d_in[2];
    const float* Wk = (const float*)d_in[3];
    const float* bk = (const float*)d_in[4];
    const float* Wv = (const float*)d_in[5];
    const float* bv = (const float*)d_in[6];
    float* out = (float*)d_out;

    const size_t QK = (size_t)NBATCH * TSEQ * HEAD; // 1,048,576 elems
    half_t* qh    = (half_t*)d_ws;
    half_t* khp   = qh + QK;                 // K in A-fragment layout
    half_t* vtp   = khp + QK;                // V^T in A-fragment layout
    half_t* wpack = vtp + QK;                // 3 * 65536 f16

    pack_w_kernel<<<(3 * WPACK_ELEMS) / 256, 256, 0, stream>>>(Wq, Wk, Wv, wpack);

    const int rowTiles = NBATCH * TSEQ / 16;  // 1024 row tiles, 8 waves/block
    qkv_proj_kernel<<<rowTiles / 8, 256, 0, stream>>>(x, wpack, bq, bk, bv,
                                                      qh, khp, vtp);

    const int waves = NBATCH * (TSEQ / 16);   // 1024 waves, 4 per block
    attn_kernel<<<waves / 4, 128, 0, stream>>>(qh, khp, vtp, out);
}
